// CausalSelfAttention_41661182771882
// MI455X (gfx1250) — matrix operand order repaired and checked by reading, not compiled, not run on hardware
//
#include <hip/hip_runtime.h>

typedef __bf16 bf16_t;
typedef __attribute__((ext_vector_type(8)))  bf16_t v8bf;
typedef __attribute__((ext_vector_type(16))) bf16_t v16bf;
typedef __attribute__((ext_vector_type(8)))  float  v8f;
typedef int v4i __attribute__((vector_size(16)));

#define B_  4
#define T_  2048
#define C_  768
#define H_  12
#define D_  64
#define C3_ 2304

// ---- async global->LDS copy (CDNA5 GLOBAL_LOAD_ASYNC_TO_LDS_B128), guarded ----
#if defined(__has_builtin)
#if __has_builtin(__builtin_amdgcn_global_load_async_to_lds_b128)
#define HAVE_ASYNC_LDS 1
#endif
#endif

typedef __attribute__((address_space(1))) v4i* gptr_v4i;
typedef __attribute__((address_space(3))) v4i* lptr_v4i;

static __device__ __forceinline__ void copy16_to_lds(const bf16_t* g, bf16_t* l) {
#ifdef HAVE_ASYNC_LDS
  __builtin_amdgcn_global_load_async_to_lds_b128((gptr_v4i)g, (lptr_v4i)l,
                                                 /*offset=*/0, /*cpol=*/0);
#else
  *(v8bf*)l = *(const v8bf*)g;
#endif
}
// Drain ASYNCcnt before the barrier; s_barrier does NOT wait on ASYNCcnt.
static __device__ __forceinline__ void async_fence() {
#ifdef HAVE_ASYNC_LDS
#if __has_builtin(__builtin_amdgcn_s_wait_asynccnt)
  __builtin_amdgcn_s_wait_asynccnt(0);
#else
  asm volatile("s_wait_asynccnt 0x0" ::: "memory");
#endif
#endif
}

// ---- WMMA fragment helpers (CDNA5 16x16x32 bf16 layouts, wave32) ----
// A (16x32): lane half h holds row m=lane%16, K = h*8+{0..7} and 16+h*8+{0..7}
// B (32x16): lane half h holds col n=lane%16, K = h*8+{0..7} and 16+h*8+{0..7}
static __device__ __forceinline__ v16bf cat16(v8bf lo, v8bf hi) {
  return __builtin_shufflevector(lo, hi, 0,1,2,3,4,5,6,7,8,9,10,11,12,13,14,15);
}
static __device__ __forceinline__ v16bf load_afrag(const bf16_t* p) {
  return cat16(*(const v8bf*)p, *(const v8bf*)(p + 16));
}
static __device__ __forceinline__ v16bf load_bfrag(const bf16_t* p) {
  return cat16(*(const v8bf*)p, *(const v8bf*)(p + 16));
}
static __device__ __forceinline__ v8f wmma_bf16(v16bf a, v16bf b, v8f c) {
  return __builtin_amdgcn_wmma_f32_16x16x32_bf16(false, a, false, b, (short)0, c,
                                                 false, false);
}

__global__ void convert_f32_to_bf16(const float* __restrict__ in,
                                    bf16_t* __restrict__ out, int n) {
  int i = blockIdx.x * blockDim.x + threadIdx.x;
  if (i < n) out[i] = (bf16_t)in[i];
}

// ============================================================================
// GEMM: C[M,N] = A[M,K] * W[N,K]^T + bias[N]; A,W bf16 row-major.
// Block = 256 threads (8 waves), block tile 128(M) x 128(N).
// Waves arranged 2(M) x 4(N); each wave: 64x32 = 4x2 WMMA tiles (8 wmma/kstep).
// A/W panels (128x32 each) double-buffered in LDS, filled with async copies.
// ============================================================================
template <int STORE_BF16>
__global__ void __launch_bounds__(256)
gemm_wmma(const bf16_t* __restrict__ A, const bf16_t* __restrict__ W,
          const float* __restrict__ bias, void* __restrict__ out,
          int K, int ldout) {
  __shared__ __align__(32) bf16_t smemA[2][128 * 32];
  __shared__ __align__(32) bf16_t smemB[2][128 * 32];

  const int tid  = threadIdx.x;
  const int lane = tid & 31;
  const int wave = tid >> 5;
  const int half = lane >> 4;
  const int ln   = lane & 15;
  const int wm   = wave >> 2;  // 0..1  -> M offset wm*64
  const int wn   = wave & 3;   // 0..3  -> N offset wn*32
  const int mb   = blockIdx.y * 128;
  const int nb   = blockIdx.x * 128;

  // Stage one 128x32 A panel + 128x32 W panel into LDS buffer `buf`.
  auto stage = [&](int buf, int k0) {
#pragma unroll
    for (int i = 0; i < 4; ++i) {
      const int c     = tid + i * 256;     // 0..1023
      const int panel = c >> 9;            // 0 = A, 1 = W
      const int idx   = c & 511;
      const int row   = idx >> 2;
      const int kc    = (idx & 3) * 8;
      const bf16_t* g = (panel ? W + (size_t)(nb + row) * K
                               : A + (size_t)(mb + row) * K) + k0 + kc;
      bf16_t* l = (panel ? smemB[buf] : smemA[buf]) + row * 32 + kc;
      copy16_to_lds(g, l);
    }
  };

  v8f acc[4][2] = {};

  stage(0, 0);
  async_fence();
  __syncthreads();

  const int nk = K >> 5;
  for (int ks = 0; ks < nk; ++ks) {
    const int cur = ks & 1;
    if (ks + 1 < nk) stage(cur ^ 1, (ks + 1) << 5);  // overlap with compute

    const bf16_t* cA = smemA[cur];
    const bf16_t* cB = smemB[cur];
    v16bf af[4], bfr[2];
#pragma unroll
    for (int i = 0; i < 4; ++i)
      af[i] = load_afrag(cA + (wm * 64 + i * 16 + ln) * 32 + half * 8);
#pragma unroll
    for (int j = 0; j < 2; ++j)
      bfr[j] = load_bfrag(cB + (wn * 32 + j * 16 + ln) * 32 + half * 8);
#pragma unroll
    for (int i = 0; i < 4; ++i)
#pragma unroll
      for (int j = 0; j < 2; ++j)
        acc[i][j] = wmma_bf16(af[i], bfr[j], acc[i][j]);

    async_fence();
    __syncthreads();
  }

#pragma unroll
  for (int j = 0; j < 2; ++j) {
    const int n  = nb + wn * 32 + j * 16 + ln;
    const float bn = bias[n];
#pragma unroll
    for (int i = 0; i < 4; ++i) {
#pragma unroll
      for (int r = 0; r < 8; ++r) {
        const int m = mb + wm * 64 + i * 16 + r + half * 8;
        const float v = acc[i][j][r] + bn;
        if (STORE_BF16) ((bf16_t*)out)[(size_t)m * ldout + n] = (bf16_t)v;
        else            ((float*)out )[(size_t)m * ldout + n] = v;
      }
    }
  }
}

// ============================================================================
// Causal flash attention: grid (T/64, H, B), block 128 (4 waves x 16 q rows).
// Key blocks of 64: S(16x64) = 8 wmma, O += P*V = 8 wmma per iteration.
// Causal masking only on the diagonal key block (uniform branch).
// qkv rows of 2304 bf16: [q(768) | k(768) | v(768)], inner [h][d].
// ============================================================================
__global__ void __launch_bounds__(128)
flash_attn(const bf16_t* __restrict__ qkv, bf16_t* __restrict__ outp) {
  __shared__ __align__(32) bf16_t vt[D_ * 64];        // V^T tile: [d][key0..63]
  __shared__ __align__(32) bf16_t pbuf[4][16 * 64];   // per-wave P (16 q x 64 k)

  const int lane = threadIdx.x & 31;
  const int wave = threadIdx.x >> 5;
  const int half = lane >> 4;
  const int ln   = lane & 15;
  const int b    = blockIdx.z;
  const int h    = blockIdx.y;
  const int qblk = blockIdx.x * 64;
  const int qbase = qblk + wave * 16;

  const size_t ldq = C3_;
  const bf16_t* qp = qkv + (size_t)b * T_ * ldq + (size_t)h * D_;
  const bf16_t* kp = qp + C_;
  const bf16_t* vp = qp + 2 * C_;

  v16bf qf0, qf1;
  {
    const bf16_t* qrow = qp + (size_t)(qbase + ln) * ldq + half * 8;
    qf0 = load_afrag(qrow);
    qf1 = load_afrag(qrow + 32);
  }

  float m_r[8], l_r[8];
#pragma unroll
  for (int r = 0; r < 8; ++r) { m_r[r] = -3.0e38f; l_r[r] = 0.0f; }
  v8f o[4] = {};
  const float scale = 0.125f;  // 1/sqrt(64)

  const int nkb = qblk / 64 + 1;  // 64-key causal blocks
  for (int kb = 0; kb < nkb; ++kb) {
    const int k0 = kb * 64;
    const bool diag = (kb == nkb - 1);

    __syncthreads();  // previous iteration's vt reads complete
    // Stage V^T (64 keys x 64 d): vectorized global reads, 2B LDS scatter.
#pragma unroll
    for (int i = 0; i < 4; ++i) {
      const int c   = threadIdx.x + i * 128;  // 0..511 chunks of 8
      const int key = c >> 3;
      const int dc  = (c & 7) * 8;
      const v8bf vv = *(const v8bf*)(vp + (size_t)(k0 + key) * ldq + dc);
#pragma unroll
      for (int e = 0; e < 8; ++e) vt[(dc + e) * 64 + key] = vv[e];
    }

    // S(16x64) = Q * K^T: 4 tiles x 2 D-chunks
    v8f s[4] = {};
#pragma unroll
    for (int t = 0; t < 4; ++t) {
      const bf16_t* krow = kp + (size_t)(k0 + t * 16 + ln) * ldq + half * 8;
      s[t] = wmma_bf16(qf0, load_bfrag(krow), s[t]);
      s[t] = wmma_bf16(qf1, load_bfrag(krow + 32), s[t]);
    }

    // Online softmax. C-layout: lane holds col n=ln (+16t), rows r+8*half.
#pragma unroll
    for (int r = 0; r < 8; ++r) {
      const int qr = qbase + r + half * 8;
      float e[4];
#pragma unroll
      for (int t = 0; t < 4; ++t) {
        float ev = s[t][r] * scale;
        if (diag && (k0 + t * 16 + ln > qr)) ev = -3.0e38f;
        e[t] = ev;
      }
      float bm = fmaxf(fmaxf(e[0], e[1]), fmaxf(e[2], e[3]));
#pragma unroll
      for (int off = 1; off < 16; off <<= 1)
        bm = fmaxf(bm, __shfl_xor(bm, off, 32));
      const float nm = fmaxf(m_r[r], bm);
      const float sc = __expf(m_r[r] - nm);
      float rs = 0.0f;
#pragma unroll
      for (int t = 0; t < 4; ++t) {
        const float p = __expf(e[t] - nm);
        rs += p;
        pbuf[wave][(r + half * 8) * 64 + t * 16 + ln] = (bf16_t)p;
      }
#pragma unroll
      for (int off = 1; off < 16; off <<= 1)
        rs += __shfl_xor(rs, off, 32);
      l_r[r] = l_r[r] * sc + rs;
      m_r[r] = nm;
#pragma unroll
      for (int nt = 0; nt < 4; ++nt) o[nt][r] *= sc;
    }

    __syncthreads();  // vt staged (pbuf is same-wave, DS in-order)

    // O(16x64) += P(16x64) * V(64x64): 4 d-tiles x 2 key-chunks
    const v16bf pa0 = load_afrag(&pbuf[wave][ln * 64 + half * 8]);
    const v16bf pa1 = load_afrag(&pbuf[wave][ln * 64 + 32 + half * 8]);
#pragma unroll
    for (int nt = 0; nt < 4; ++nt) {
      const bf16_t* vr = &vt[(nt * 16 + ln) * 64 + half * 8];
      o[nt] = wmma_bf16(pa0, load_bfrag(vr), o[nt]);
      o[nt] = wmma_bf16(pa1, load_bfrag(vr + 32), o[nt]);
    }
  }

  bf16_t* op = outp + (size_t)b * T_ * C_ + (size_t)h * D_;
#pragma unroll
  for (int r = 0; r < 8; ++r) {
    const int m = qbase + r + half * 8;
    const float inv = 1.0f / l_r[r];
#pragma unroll
    for (int nt = 0; nt < 4; ++nt)
      op[(size_t)m * C_ + nt * 16 + ln] = (bf16_t)(o[nt][r] * inv);
  }
}

extern "C" void kernel_launch(void* const* d_in, const int* in_sizes, int n_in,
                              void* d_out, int out_size, void* d_ws, size_t ws_size,
                              hipStream_t stream) {
  (void)in_sizes; (void)n_in; (void)out_size; (void)ws_size;
  const float* x      = (const float*)d_in[0];
  const float* w_qkv  = (const float*)d_in[1];
  const float* b_qkv  = (const float*)d_in[2];
  const float* w_proj = (const float*)d_in[3];
  const float* b_proj = (const float*)d_in[4];
  float* out = (float*)d_out;

  char* ws = (char*)d_ws;
  size_t off = 0;
  auto take = [&](size_t elems) -> bf16_t* {
    bf16_t* p = (bf16_t*)(ws + off);
    off = (off + elems * sizeof(bf16_t) + 255) & ~(size_t)255;
    return p;
  };
  const size_t nX   = (size_t)B_ * T_ * C_;
  const size_t nWq  = (size_t)C3_ * C_;
  const size_t nWp  = (size_t)C_ * C_;
  const size_t nQKV = (size_t)B_ * T_ * C3_;
  bf16_t* xb   = take(nX);
  bf16_t* wqb  = take(nWq);
  bf16_t* wpb  = take(nWp);
  bf16_t* qkvb = take(nQKV);
  bf16_t* atb  = take(nX);

  convert_f32_to_bf16<<<(int)((nX  + 255) / 256), 256, 0, stream>>>(x,      xb,  (int)nX);
  convert_f32_to_bf16<<<(int)((nWq + 255) / 256), 256, 0, stream>>>(w_qkv,  wqb, (int)nWq);
  convert_f32_to_bf16<<<(int)((nWp + 255) / 256), 256, 0, stream>>>(w_proj, wpb, (int)nWp);

  // QKV projection: [8192x768] x [2304x768]^T + b -> bf16 [8192x2304]
  gemm_wmma<1><<<dim3(C3_ / 128, (B_ * T_) / 128), 256, 0, stream>>>(
      xb, wqb, b_qkv, qkvb, C_, C3_);

  // Causal flash attention -> bf16 [8192x768]
  flash_attn<<<dim3(T_ / 64, H_, B_), 128, 0, stream>>>(qkvb, atb);

  // Output projection: [8192x768] x [768x768]^T + b -> f32 d_out
  gemm_wmma<0><<<dim3(C_ / 128, (B_ * T_) / 128), 256, 0, stream>>>(
      atb, wpb, b_proj, out, C_, C_);
}
